// CGCNNConv_27908697489524
// MI455X (gfx1250) — matrix operand order repaired and checked by reading, not compiled, not run on hardware
//
#include <hip/hip_runtime.h>

typedef __bf16    v16bf __attribute__((ext_vector_type(16)));
typedef float     v8f   __attribute__((ext_vector_type(8)));
typedef unsigned  u32x4 __attribute__((ext_vector_type(4)));

// LDS (addrspace 3) volatile pointer: guarantees ds_load_* and blocks LICM hoist
typedef __attribute__((address_space(3))) const volatile u32x4* lds_cvq_ptr;

#define FEA 64
#define NBR 12
#define KDIM 192
#define ODIM 128

// pack two f32 -> two bf16 (round-half-up) in ONE v_perm_b32 + 2 adds
__device__ __forceinline__ unsigned pack2bf(float a, float b) {
    unsigned ua = __float_as_uint(a) + 0x8000u;
    unsigned ub = __float_as_uint(b) + 0x8000u;
    // out bytes [0,1] = ua bytes [2,3]; out bytes [2,3] = ub bytes [2,3]
    return __builtin_amdgcn_perm(ub, ua, 0x07060302u);
}
__device__ __forceinline__ float softplus_f(float x) {
    return (x > 20.f) ? x : __logf(1.f + __expf(x));
}

union V16 { unsigned u[8]; v16bf v; };
union BFrag { u32x4 q[2]; v16bf v; };

// build one 16x32 bf16 A-fragment for k-step kk from the three concat sources
__device__ __forceinline__ v16bf build_afrag(const float* __restrict__ srcA,
                                             const float* __restrict__ srcB,
                                             const float* __restrict__ srcC,
                                             int kk, int laneHi) {
    const float* src = (kk < 2) ? srcA : (kk < 4) ? srcB : srcC;
    const int off = kk * 32 - ((kk < 2) ? 0 : (kk < 4) ? 64 : 128) + laneHi * 8;
    const float4* p = (const float4*)(src + off);
    float4 x0 = p[0], x1 = p[1];   // K = base+0..7   (v0..v3)
    float4 y0 = p[4], y1 = p[5];   // K = base+16..23 (v4..v7)
    V16 af;
    af.u[0] = pack2bf(x0.x, x0.y); af.u[1] = pack2bf(x0.z, x0.w);
    af.u[2] = pack2bf(x1.x, x1.y); af.u[3] = pack2bf(x1.z, x1.w);
    af.u[4] = pack2bf(y0.x, y0.y); af.u[5] = pack2bf(y0.z, y0.w);
    af.u[6] = pack2bf(y1.x, y1.y); af.u[7] = pack2bf(y1.z, y1.w);
    return af.v;
}

// ---------------------------------------------------------------------------
// Fused gather+concat+GEMM (bf16 WMMA, f32 accum). Each wave owns a pair of
// 16-row tiles (32 rows) so every LDS B-fragment load feeds two WMMAs.
// PASS 1: accumulate per-channel sum / sumsq for BN1.
// PASS 2: apply BN1 affine, sigmoid*softplus, neighbor-sum into msg (atomics).
// ---------------------------------------------------------------------------
template <int PASS>
__global__ __launch_bounds__(256, 4)
void cgcnn_gemm(const float* __restrict__ atom_fea,
                const float* __restrict__ nbr_fea,
                const long long* __restrict__ nbr_idx,
                const float* __restrict__ W,      // [192,128]
                const float* __restrict__ bias,   // [128]
                float* __restrict__ stats1,       // pass1: [256] sum|sumsq
                const float* __restrict__ scale1, // pass2: [128]
                const float* __restrict__ shift1, // pass2: [128]
                float* __restrict__ msg,          // pass2: [natoms*64]
                int natoms) {
    // B fragments pre-swizzled in LDS: 6 k-steps * 8 n-tiles, 32 lanes * 8 dwords
    __shared__ __attribute__((aligned(64))) unsigned wlds[6 * 8 * 32 * 8];

    const int tid = threadIdx.x;
    // ---- convert W to bf16, pre-swizzled into B-fragment layout ----
    // B 32x16 bf16 layout: lanes 0-15 -> K 0..7,16..23 (v0..v7), lanes 16-31 -> K 8..15,24..31
    for (int i = tid; i < 12288; i += blockDim.x) {
        int b   = i >> 8;            // fragment block (kk*8+t)
        int rem = i & 255;
        int ln  = rem >> 3;          // lane 0..31
        int d   = rem & 7;           // dword 0..7
        int kk  = b >> 3, t = b & 7;
        int n   = t * 16 + (ln & 15);
        int K   = kk * 32 + (ln >> 4) * 8 + (d >> 2) * 16 + (d & 3) * 2;
        wlds[i] = pack2bf(W[(size_t)K * ODIM + n], W[(size_t)(K + 1) * ODIM + n]);
    }
    __syncthreads();

    // AS(3) volatile view of the weight fragments: low 32 bits of a generic
    // LDS address are the LDS byte offset -> reinterpret as LDS pointer.
    lds_cvq_ptr wq = (lds_cvq_ptr)(unsigned)(uintptr_t)&wlds[0];

    const int lane    = tid & 31;
    const int laneMod = lane & 15;
    const int laneHi  = lane >> 4;
    const int wave    = (blockIdx.x * blockDim.x + tid) >> 5;
    const int nwaves  = (gridDim.x * blockDim.x) >> 5;
    const int rows    = natoms * NBR;
    const int npairs  = rows >> 5;   // 32 rows per pair (1.2M/32 = 37500)

    float bia[8], sc1[8], sh1[8];
#pragma unroll
    for (int t = 0; t < 8; ++t) {
        int ch = t * 16 + laneMod;
        bia[t] = bias[ch];
        if (PASS == 2) { sc1[t] = scale1[ch]; sh1[t] = shift1[ch]; }
        else           { sc1[t] = 0.f;        sh1[t] = 0.f;        }
    }
    float runS[8], runQ[8];
#pragma unroll
    for (int t = 0; t < 8; ++t) { runS[t] = 0.f; runQ[t] = 0.f; }
    int nsub = 0;   // subtiles processed (for bias fixup in pass 1)

    for (int pr = wave; pr < npairs; pr += nwaves) {
        const int rowbase = pr << 5;
        const int r0 = rowbase + laneMod;        // A-row, subtile 0
        const int r1 = r0 + 16;                  // A-row, subtile 1
        const float* sA0 = atom_fea + (size_t)(r0 / NBR) * FEA;
        const float* sB0 = atom_fea + (size_t)nbr_idx[r0] * FEA;
        const float* sC0 = nbr_fea  + (size_t)r0 * FEA;
        const float* sA1 = atom_fea + (size_t)(r1 / NBR) * FEA;
        const float* sB1 = atom_fea + (size_t)nbr_idx[r1] * FEA;
        const float* sC1 = nbr_fea  + (size_t)r1 * FEA;

        // prefetch the streaming operand + indices for this wave's next pair
        const int nextRow = r0 + (nwaves << 5);
        if (nextRow < rows) {
            __builtin_prefetch(nbr_fea + (size_t)nextRow * FEA, 0, 3);
            __builtin_prefetch(nbr_fea + (size_t)(nextRow + 16) * FEA, 0, 3);
            __builtin_prefetch(nbr_idx + nextRow, 0, 3);
        }

        v8f acc0[8], acc1[8];
#pragma unroll
        for (int t = 0; t < 8; ++t) { acc0[t] = {}; acc1[t] = {}; }

#pragma unroll
        for (int kk = 0; kk < 6; ++kk) {
            const v16bf af0 = build_afrag(sA0, sB0, sC0, kk, laneHi);
            const v16bf af1 = build_afrag(sA1, sB1, sC1, kk, laneHi);
#pragma unroll
            for (int t = 0; t < 8; ++t) {
                BFrag b;
                const int base = ((kk * 8 + t) << 6) + (lane << 1);
                b.q[0] = wq[base];
                b.q[1] = wq[base + 1];
                acc0[t] = __builtin_amdgcn_wmma_f32_16x16x32_bf16(
                    false, af0, false, b.v, (short)0, acc0[t], false, false);
                acc1[t] = __builtin_amdgcn_wmma_f32_16x16x32_bf16(
                    false, af1, false, b.v, (short)0, acc1[t], false, false);
            }
        }

        if (PASS == 1) {
            // raw sums; bias folded in algebraically at the end
#pragma unroll
            for (int t = 0; t < 8; ++t)
#pragma unroll
                for (int v = 0; v < 8; ++v) {
                    float x0 = acc0[t][v], x1 = acc1[t][v];
                    runS[t] += x0 + x1;
                    runQ[t] += x0 * x0 + x1 * x1;
                }
            nsub += 2;
        } else {
            // C/D layout: VGPR v -> row M = v + 8*laneHi, col N = laneMod
#pragma unroll
            for (int s = 0; s < 2; ++s) {
                const v8f* acc = s ? acc1 : acc0;
                const int rb = rowbase + s * 16;
                int atomv[8];
#pragma unroll
                for (int v = 0; v < 8; ++v)
                    atomv[v] = (rb + v + laneHi * 8) / NBR;
#pragma unroll
                for (int t = 0; t < 4; ++t) {
                    float pend = 0.f;
                    int curAtom = -1;
#pragma unroll
                    for (int v = 0; v < 8; ++v) {
                        float f = (acc[t][v]     + bia[t])     * sc1[t]     + sh1[t];
                        float c = (acc[t + 4][v] + bia[t + 4]) * sc1[t + 4] + sh1[t + 4];
                        float g = (1.f / (1.f + __expf(-f))) * softplus_f(c);
                        int a = atomv[v];
                        if (a != curAtom) {   // combine consecutive rows of same atom
                            if (curAtom >= 0)
                                atomicAdd(&msg[(size_t)curAtom * FEA + t * 16 + laneMod], pend);
                            curAtom = a; pend = 0.f;
                        }
                        pend += g;
                    }
                    atomicAdd(&msg[(size_t)curAtom * FEA + t * 16 + laneMod], pend);
                }
            }
        }
    }

    if (PASS == 1) {
        const float cnt = 8.0f * (float)nsub;   // elements per lane per t
#pragma unroll
        for (int t = 0; t < 8; ++t) {
            int ch = t * 16 + laneMod;
            float b = bia[t];
            float S = runS[t] + cnt * b;                         // sum(x+b)
            float Q = runQ[t] + 2.f * b * runS[t] + cnt * b * b; // sum((x+b)^2)
            atomicAdd(&stats1[ch], S);          // both lane halves hold disjoint rows
            atomicAdd(&stats1[ODIM + ch], Q);
        }
    }
}

// ---------------------------------------------------------------------------
__global__ void finalize_bn(const float* __restrict__ stats,
                            const float* __restrict__ gamma,
                            const float* __restrict__ beta,
                            float* __restrict__ scale, float* __restrict__ shift,
                            int nch, float invcnt) {
    int i = blockIdx.x * blockDim.x + threadIdx.x;
    if (i < nch) {
        float mean = stats[i] * invcnt;
        float var  = stats[nch + i] * invcnt - mean * mean;
        float sc   = gamma[i] * rsqrtf(var + 1e-5f);
        scale[i] = sc;
        shift[i] = beta[i] - mean * sc;
    }
}

__global__ __launch_bounds__(256)
void bn2_stats(const float* __restrict__ msg, float* __restrict__ stats2, int natoms) {
    __shared__ float redS[256], redQ[256];
    int f = threadIdx.x & 63, sub = threadIdx.x >> 6;
    float s = 0.f, q = 0.f;
    for (int n = blockIdx.x * 4 + sub; n < natoms; n += gridDim.x * 4) {
        float v = msg[(size_t)n * FEA + f];
        s += v; q += v * v;
    }
    redS[threadIdx.x] = s; redQ[threadIdx.x] = q;
    __syncthreads();
    if (sub == 0) {
        s = redS[f] + redS[64 + f] + redS[128 + f] + redS[192 + f];
        q = redQ[f] + redQ[64 + f] + redQ[128 + f] + redQ[192 + f];
        atomicAdd(&stats2[f], s);
        atomicAdd(&stats2[FEA + f], q);
    }
}

__global__ void final_out(const float* __restrict__ atom_fea,
                          const float* __restrict__ msg,
                          const float* __restrict__ scale2,
                          const float* __restrict__ shift2,
                          float* __restrict__ out, int total) {
    int i = blockIdx.x * blockDim.x + threadIdx.x;
    if (i < total) {
        int f = i & 63;
        float x = msg[i] * scale2[f] + shift2[f];
        out[i] = atom_fea[i] + softplus_f(x);
    }
}

// ---------------------------------------------------------------------------
extern "C" void kernel_launch(void* const* d_in, const int* in_sizes, int n_in,
                              void* d_out, int out_size, void* d_ws, size_t ws_size,
                              hipStream_t stream) {
    (void)n_in; (void)out_size; (void)ws_size;
    const float*     atom_fea = (const float*)d_in[0];
    const float*     nbr_fea  = (const float*)d_in[1];
    const long long* nbr_idx  = (const long long*)d_in[2];
    const float*     W        = (const float*)d_in[3];
    const float*     bias     = (const float*)d_in[4];
    const float*     g1       = (const float*)d_in[5];
    const float*     b1       = (const float*)d_in[6];
    const float*     g2       = (const float*)d_in[7];
    const float*     b2       = (const float*)d_in[8];
    float*           out      = (float*)d_out;

    const int natoms = in_sizes[0] / FEA;
    const int rows   = natoms * NBR;

    float* ws     = (float*)d_ws;
    float* stats1 = ws;          // 256
    float* scale1 = ws + 256;    // 128
    float* shift1 = ws + 384;    // 128
    float* stats2 = ws + 512;    // 128
    float* scale2 = ws + 640;    // 64
    float* shift2 = ws + 704;    // 64
    float* msg    = ws + 768;    // natoms*64

    hipMemsetAsync(stats1, 0, 768 * sizeof(float), stream);
    hipMemsetAsync(msg, 0, (size_t)natoms * FEA * sizeof(float), stream);

    dim3 grid(1024), block(256);
    cgcnn_gemm<1><<<grid, block, 0, stream>>>(atom_fea, nbr_fea, nbr_idx, W, bias,
                                              stats1, nullptr, nullptr, nullptr, natoms);
    finalize_bn<<<1, 128, 0, stream>>>(stats1, g1, b1, scale1, shift1, ODIM,
                                       1.0f / (float)rows);
    cgcnn_gemm<2><<<grid, block, 0, stream>>>(atom_fea, nbr_fea, nbr_idx, W, bias,
                                              nullptr, scale1, shift1, msg, natoms);
    bn2_stats<<<256, 256, 0, stream>>>(msg, stats2, natoms);
    finalize_bn<<<1, 64, 0, stream>>>(stats2, g2, b2, scale2, shift2, FEA,
                                      1.0f / (float)natoms);
    const int total = natoms * FEA;
    final_out<<<(total + 255) / 256, 256, 0, stream>>>(atom_fea, msg, scale2, shift2,
                                                       out, total);
}
